// DeepAAIKmerEmbeddingCls_74741020885018
// MI455X (gfx1250) — matrix operand order repaired and checked by reading, not compiled, not run on hardware
//
#include <hip/hip_runtime.h>
#include <math.h>

// ---------------- model dimensions ----------------
#define Hh    512
#define KMERD 8420
#define NAB   6000
#define NV    3000
#define BB    32
#define LAB   344
#define LV    912

typedef __attribute__((ext_vector_type(16))) __bf16 v16bf;
typedef __attribute__((ext_vector_type(8)))  float  v8f;

__device__ __forceinline__ unsigned bfbits(float f) {
  union { float f; unsigned u; } a; a.f = f;
  return (a.u + 0x7FFFu + ((a.u >> 16) & 1u)) >> 16;   // RNE to bf16 bits
}
__device__ __forceinline__ __bf16 f2bf(float f) {
  unsigned short s = (unsigned short)bfbits(f);
  union { unsigned short s; __bf16 b; } o; o.s = s;
  return o.b;
}
__device__ __forceinline__ unsigned pk2(float lo, float hi) {
  return bfbits(lo) | (bfbits(hi) << 16);
}
__device__ __forceinline__ float eluf(float x) { return x > 0.f ? x : (expf(x) - 1.f); }
__device__ __forceinline__ float sigf(float x) { return 1.f / (1.f + expf(-x)); }
__device__ __forceinline__ int   imin(int a, int b) { return a < b ? a : b; }
__device__ __forceinline__ int   iclamp(int v, int lo, int hi) {
  return v < lo ? lo : (v > hi ? hi : v);
}

union FragAB { v16bf v; uint4 q[2]; };

// ---------------------------------------------------------------------------
// WMMA bf16->f32 GEMM:  C = epilogue(A[MxK] @ B + bias)
//   BT=0: B is [K,N] row-major (requires N % 64 == 0 — true for all callers).
//   BT=1: B is [N,K] row-major (A @ B^T).
//   ADDRES: C += R.  COSDIV: C /= (wv[m]*wv[n]).  ACT: 0 none, 1 ELU, 2 tanh.
// Block tile 128x64, KT=32; 256 threads = 8 waves; each wave owns a 32x32
// output patch -> 4 v_wmma_f32_16x16x32_bf16 per K-step.
// Main K-loop is fully branch-free: OOB rows/cols are *clamped* (their
// garbage only reaches unstored outputs). The ragged K tail zero-fills each
// thread's private LDS chunk and copies only the valid contiguous elements
// with a runtime-count loop (no per-element clamps, no exec divergence).
// ---------------------------------------------------------------------------
template<int ACT, int ADDRES, int BT, int COSDIV>
__global__ __launch_bounds__(256)
void gemm_wmma(const float* __restrict__ A, const float* __restrict__ B,
               const float* __restrict__ bias, const float* __restrict__ R,
               const float* __restrict__ wv, float* __restrict__ C,
               int M, int N, int K, int lda, int ldb, int ldc)
{
  __shared__ __align__(16) __bf16 As[128][40];  // [m][k], 80B rows
  __shared__ __align__(16) __bf16 Bs[64][40];   // [n][k] (transposed)

  const int tid  = threadIdx.x;
  const int lane = tid & 31;
  const int wave = tid >> 5;
  const int wm   = wave & 3;      // 32-row subtile
  const int wn   = wave >> 2;     // 32-col half

  const int m0 = blockIdx.y * 128;
  const int n0 = blockIdx.x * 64;

  v8f acc[2][2];
#pragma unroll
  for (int i = 0; i < 2; ++i)
#pragma unroll
    for (int j = 0; j < 2; ++j)
      acc[i][j] = (v8f){0.f,0.f,0.f,0.f,0.f,0.f,0.f,0.f};

  // fragment addressing (ISA 16-bit A 16x32 / B 32x16 layouts)
  const int arow0 = wm * 32 + (lane & 15);
  const int arow1 = arow0 + 16;
  const int abase = (lane >= 16) ? 8 : 0;
  const int bbase = (lane >= 16) ? 8 : 0;
  const int bn0   = wn * 32 + (lane & 15);
  const int bn1   = bn0 + 16;

  // ---- staging coordinates (clamped; no predication) ----
  const int sa_row = tid >> 1;                   // 0..127
  const int sa_col = (tid & 1) * 16;             // 0 or 16
  const int gma    = imin(m0 + sa_row, M - 1);   // clamp A row
  const float* pa  = A + (size_t)gma * lda + sa_col;

  int sb_n, sb_k, gnb = 0, nbc = 0;
  const float* pb;
  if (BT) {
    sb_n = tid >> 2;                 // 0..63
    sb_k = (tid & 3) * 8;            // 0,8,16,24
    gnb  = imin(n0 + sb_n, N - 1);   // clamp B row (= output col)
    pb   = B + (size_t)gnb * ldb + sb_k;
  } else {
    sb_k = (tid & 15) * 2;           // even k pair base 0..30
    sb_n = (tid >> 4) * 4;           // 0..60
    nbc  = n0 + sb_n;                // callers guarantee N % 64 == 0
    pb   = B + (size_t)sb_k * ldb + nbc;
  }

  const int Kmain = K & ~31;

  auto mma_step = [&]() {
    FragAB a0, a1, b0, b1;
    a0.q[0] = *(const uint4*)&As[arow0][abase];
    a0.q[1] = *(const uint4*)&As[arow0][abase + 16];
    a1.q[0] = *(const uint4*)&As[arow1][abase];
    a1.q[1] = *(const uint4*)&As[arow1][abase + 16];
    b0.q[0] = *(const uint4*)&Bs[bn0][bbase];
    b0.q[1] = *(const uint4*)&Bs[bn0][bbase + 16];
    b1.q[0] = *(const uint4*)&Bs[bn1][bbase];
    b1.q[1] = *(const uint4*)&Bs[bn1][bbase + 16];
    acc[0][0] = __builtin_amdgcn_wmma_f32_16x16x32_bf16(false, a0.v, false, b0.v,
                                                        (short)0, acc[0][0], false, false);
    acc[0][1] = __builtin_amdgcn_wmma_f32_16x16x32_bf16(false, a0.v, false, b1.v,
                                                        (short)0, acc[0][1], false, false);
    acc[1][0] = __builtin_amdgcn_wmma_f32_16x16x32_bf16(false, a1.v, false, b0.v,
                                                        (short)0, acc[1][0], false, false);
    acc[1][1] = __builtin_amdgcn_wmma_f32_16x16x32_bf16(false, a1.v, false, b1.v,
                                                        (short)0, acc[1][1], false, false);
  };

  // ---------------- branch-free main K loop ----------------
  for (int k0 = 0; k0 < Kmain; k0 += 32) {
    __syncthreads();
    { // A: 128x32, 16 floats/thread -> 2x ds_store_b128
      float4 f0 = *(const float4*)(pa + 0);
      float4 f1 = *(const float4*)(pa + 4);
      float4 f2 = *(const float4*)(pa + 8);
      float4 f3 = *(const float4*)(pa + 12);
      uint4 p0 = { pk2(f0.x,f0.y), pk2(f0.z,f0.w), pk2(f1.x,f1.y), pk2(f1.z,f1.w) };
      uint4 p1 = { pk2(f2.x,f2.y), pk2(f2.z,f2.w), pk2(f3.x,f3.y), pk2(f3.z,f3.w) };
      *(uint4*)&As[sa_row][sa_col]     = p0;
      *(uint4*)&As[sa_row][sa_col + 8] = p1;
      __builtin_prefetch(pa + 32, 0, 1);
      pa += 32;
    }
    if (BT) { // B rows contiguous in K
      float4 f0 = *(const float4*)(pb + 0);
      float4 f1 = *(const float4*)(pb + 4);
      uint4 p = { pk2(f0.x,f0.y), pk2(f0.z,f0.w), pk2(f1.x,f1.y), pk2(f1.z,f1.w) };
      *(uint4*)&Bs[sb_n][sb_k] = p;
      __builtin_prefetch(pb + 32, 0, 1);
      pb += 32;
    } else {  // B [K,N]: two adjacent K rows x 4 cols -> 4 packed ds_store_b32
      float4 f0 = *(const float4*)pb;
      float4 f1 = *(const float4*)(pb + ldb);
      *(unsigned*)&Bs[sb_n + 0][sb_k] = pk2(f0.x, f1.x);
      *(unsigned*)&Bs[sb_n + 1][sb_k] = pk2(f0.y, f1.y);
      *(unsigned*)&Bs[sb_n + 2][sb_k] = pk2(f0.z, f1.z);
      *(unsigned*)&Bs[sb_n + 3][sb_k] = pk2(f0.w, f1.w);
      __builtin_prefetch(pb + (size_t)32 * ldb, 0, 1);
      pb += (size_t)32 * ldb;
    }
    __syncthreads();
    mma_step();
  }

  // ------ single masked K tail: zero own chunk, copy valid prefix ------
  if (Kmain < K) {
    __syncthreads();
    const uint4 z = {0u, 0u, 0u, 0u};
    { // A tail (each thread owns As[sa_row][sa_col..+15])
      *(uint4*)&As[sa_row][sa_col]     = z;
      *(uint4*)&As[sa_row][sa_col + 8] = z;
      int cnt = iclamp(K - (Kmain + sa_col), 0, 16);
      for (int e = 0; e < cnt; ++e)
        As[sa_row][sa_col + e] = f2bf(pa[e]);     // pa already at Kmain+sa_col
    }
    if (BT) { // each thread owns Bs[sb_n][sb_k..+7]
      *(uint4*)&Bs[sb_n][sb_k] = z;
      int cnt = iclamp(K - (Kmain + sb_k), 0, 8);
      for (int e = 0; e < cnt; ++e)
        Bs[sb_n][sb_k + e] = f2bf(pb[e]);         // pb already at row gnb, Kmain+sb_k
    } else {  // each thread owns Bs[sb_n..+3][sb_k..+1]
      *(unsigned*)&Bs[sb_n + 0][sb_k] = 0u;
      *(unsigned*)&Bs[sb_n + 1][sb_k] = 0u;
      *(unsigned*)&Bs[sb_n + 2][sb_k] = 0u;
      *(unsigned*)&Bs[sb_n + 3][sb_k] = 0u;
      int rows = iclamp(K - (Kmain + sb_k), 0, 2);
      const float* q = pb;                         // pb already at row Kmain+sb_k
      for (int r = 0; r < rows; ++r) {
        float4 f = *(const float4*)q;
        Bs[sb_n + 0][sb_k + r] = f2bf(f.x);
        Bs[sb_n + 1][sb_k + r] = f2bf(f.y);
        Bs[sb_n + 2][sb_k + r] = f2bf(f.z);
        Bs[sb_n + 3][sb_k + r] = f2bf(f.w);
        q += ldb;
      }
    }
    __syncthreads();
    mma_step();
  }

  // ---------------- epilogue ----------------
  const bool interior = (m0 + 128 <= M) && (n0 + 64 <= N);
#pragma unroll
  for (int i = 0; i < 2; ++i) {
#pragma unroll
    for (int j = 0; j < 2; ++j) {
#pragma unroll
      for (int r = 0; r < 8; ++r) {
        int mm = wm * 32 + i * 16 + ((lane < 16) ? r : r + 8);
        int nn = wn * 32 + j * 16 + (lane & 15);
        int gm = m0 + mm, gn = n0 + nn;
        if (interior || (gm < M && gn < N)) {
          float v = acc[i][j][r];
          if (bias)   v += bias[gn];
          if (ADDRES) v += R[(size_t)gm * ldc + gn];
          if (COSDIV) v  = v / (wv[gm] * wv[gn]);
          if (ACT == 1) v = eluf(v);
          if (ACT == 2) v = tanhf(v);
          C[(size_t)gm * ldc + gn] = v;
        }
      }
    }
  }
}

// ---------------- small helper kernels ----------------
__global__ void k_elu(const float* __restrict__ x, float* __restrict__ y, int n) {
  int i = blockIdx.x * 256 + threadIdx.x;
  if (i < n) y[i] = eluf(x[i]);
}

__global__ __launch_bounds__(256)
void k_rownorm(const float* __restrict__ t, float* __restrict__ w, int rows) {
  int wave = threadIdx.x >> 5, lane = threadIdx.x & 31;
  int row = blockIdx.x * 8 + wave;
  if (row >= rows) return;
  const float* p = t + (size_t)row * Hh;
  float s = 0.f;
  for (int j = lane; j < Hh; j += 32) { float v = p[j]; s += v * v; }
#pragma unroll
  for (int o = 16; o > 0; o >>= 1) s += __shfl_xor(s, o, 32);
  if (lane == 0) w[row] = sqrtf(s);
}

__global__ void k_gather_elu(const float* __restrict__ res, const int* __restrict__ idx,
                             float* __restrict__ out) {
  int b = blockIdx.x, j = threadIdx.x;          // grid 32, block 512
  int r = idx[b];
  out[b * Hh + j] = eluf(res[(size_t)r * Hh + j]);
}

// conv1d(k=2, pad=1) on [B,L,22] + maxpool(k=2,s=1) + bias,
// writing channel-major flattened [B, 64*L] for the following linear.
__device__ __forceinline__ float conv_at(const float* __restrict__ x,
                                         const float* __restrict__ cw, int L, int p) {
  float s = 0.f;
  const float* x0 = x + (size_t)(p - 1) * 22;
#pragma unroll
  for (int i = 0; i < 22; ++i) {
    float a0 = (p - 1 >= 0) ? x0[i]       : 0.f;
    float a1 = (p     <  L) ? x0[22 + i]  : 0.f;
    s += a0 * cw[i * 2] + a1 * cw[i * 2 + 1];
  }
  return s;
}
__global__ void k_cnn(const float* __restrict__ amino, const float* __restrict__ cW,
                      const float* __restrict__ cb, float* __restrict__ out, int L) {
  int b = blockIdx.y;
  int e = blockIdx.x * 256 + threadIdx.x;
  int total = 64 * L;
  if (e >= total) return;
  int c = e / L, l = e % L;
  const float* x  = amino + (size_t)b * L * 22;
  const float* cw = cW + c * 44;
  float v0 = conv_at(x, cw, L, l);
  float v1 = conv_at(x, cw, L, l + 1);
  out[(size_t)b * total + e] = fmaxf(v0, v1) + cb[c];   // e == c*L + l
}

// two-stage 1x1-conv channel attention over [B,2,512]; x=global, r=local
__global__ void k_att(const float* __restrict__ vres, const float* __restrict__ abres,
                      const float* __restrict__ vft,  const float* __restrict__ abft,
                      const float* __restrict__ W1, const float* __restrict__ b1,
                      const float* __restrict__ W2, const float* __restrict__ b2,
                      float* __restrict__ pair) {
  int b = blockIdx.x, l = threadIdx.x;            // grid 32, block 512
  float x0 = vres[b * Hh + l],  x1 = abres[b * Hh + l];
  float r0 = vft[b * Hh + l],   r1 = abft[b * Hh + l];
  float s0 = x0 + r0, s1 = x1 + r1;
  float w0 = sigf(W1[0] * s0 + W1[1] * s1 + b1[0]);
  float w1 = sigf(W1[2] * s0 + W1[3] * s1 + b1[1]);
  float xi0 = x0 * w0 + r0 * (1.f - w0);
  float xi1 = x1 * w1 + r1 * (1.f - w1);
  float u0 = sigf(W2[0] * xi0 + W2[1] * xi1 + b2[0]);
  float u1 = sigf(W2[2] * xi0 + W2[3] * xi1 + b2[1]);
  pair[b * 1024 + l]       = x0 * u0 + r0 * (1.f - u0);
  pair[b * 1024 + 512 + l] = x1 * u1 + r1 * (1.f - u1);
}

__global__ void k_pred(const float* __restrict__ ft, const float* __restrict__ pW,
                       const float* __restrict__ pb, float* __restrict__ out) {
  __shared__ float red[512];
  int b = blockIdx.x, t = threadIdx.x;            // grid 32, block 512
  red[t] = ft[b * Hh + t] * pW[t];
  __syncthreads();
  for (int s = 256; s > 0; s >>= 1) { if (t < s) red[t] += red[t + s]; __syncthreads(); }
  if (t == 0) out[b] = sigf(red[0] + pb[0]);
}

// ---------------------------------------------------------------------------
extern "C" void kernel_launch(void* const* d_in, const int* in_sizes, int n_in,
                              void* d_out, int out_size, void* d_ws, size_t ws_size,
                              hipStream_t stream) {
  const float* ab_kmer = (const float*)d_in[0];
  const float* v_kmer  = (const float*)d_in[1];
  const float* ab_am   = (const float*)d_in[2];
  const float* v_am    = (const float*)d_in[3];
  const int*   ab_idx  = (const int*)d_in[4];
  const int*   v_idx   = (const int*)d_in[5];
  const float* abkW = (const float*)d_in[6];  const float* abkb = (const float*)d_in[7];
  const float* vkW  = (const float*)d_in[8];  const float* vkb  = (const float*)d_in[9];
  const float* shW  = (const float*)d_in[10]; const float* shb  = (const float*)d_in[11];
  const float* abaW = (const float*)d_in[12]; const float* abab = (const float*)d_in[13];
  const float* vaW  = (const float*)d_in[14]; const float* vab  = (const float*)d_in[15];
  const float* g1W  = (const float*)d_in[16]; const float* g1b  = (const float*)d_in[17];
  const float* g2W  = (const float*)d_in[18]; const float* g2b  = (const float*)d_in[19];
  const float* c1cW = (const float*)d_in[20]; const float* c1cb = (const float*)d_in[21];
  const float* c1lW = (const float*)d_in[22]; const float* c1lb = (const float*)d_in[23];
  const float* c2cW = (const float*)d_in[24]; const float* c2cb = (const float*)d_in[25];
  const float* c2lW = (const float*)d_in[26]; const float* c2lb = (const float*)d_in[27];
  const float* a1W  = (const float*)d_in[28]; const float* a1b  = (const float*)d_in[29];
  const float* a2W  = (const float*)d_in[30]; const float* a2b  = (const float*)d_in[31];
  const float* oW   = (const float*)d_in[32]; const float* ob   = (const float*)d_in[33];
  const float* pW   = (const float*)d_in[34]; const float* pb   = (const float*)d_in[35];

  float* out = (float*)d_out;
  float* ab_adj = out + 32;                               // [6000,6000]
  float* v_adj  = ab_adj + (size_t)NAB * NAB;             // [3000,3000]

  // workspace layout (f32 elements); buffers reused serially across branches
  float* X    = (float*)d_ws;                             // [6000,512]
  float* RES  = X    + (size_t)NAB * Hh;                  // [6000,512]
  float* T    = RES  + (size_t)NAB * Hh;                  // [6000,512]
  float* Wn   = T    + (size_t)NAB * Hh;                  // [6016]
  float* AB32 = Wn   + 6016;                              // [32,512]
  float* V32  = AB32 + BB * Hh;                           // [32,512]
  float* C1   = V32  + BB * Hh;                           // [32, 64*344]
  float* C2   = C1   + (size_t)BB * 64 * LAB;             // [32, 64*912]
  float* ABFT = C2   + (size_t)BB * 64 * LV;              // [32,512]
  float* VFT  = ABFT + BB * Hh;                           // [32,512]
  float* PAIR = VFT  + BB * Hh;                           // [32,1024]
  float* OUTF = PAIR + BB * 1024;                         // [32,512]

  auto ntiles = [](int n) { return (n + 63) / 64; };
  auto mtiles = [](int m) { return (m + 127) / 128; };

  // ---------------- graph branch (ab then v, reusing X/RES/T/Wn) ----------
  struct Branch { const float* kmer; const float* kW; const float* kb;
                  const float* aW; const float* ab; const int* idx;
                  float* adj; float* res32; int M; };
  Branch br[2] = {
    { ab_kmer, abkW, abkb, abaW, abab, ab_idx, ab_adj, AB32, NAB },
    { v_kmer,  vkW,  vkb,  vaW,  vab,  v_idx,  v_adj,  V32,  NV  },
  };

  for (int s = 0; s < 2; ++s) {
    const Branch& Bx = br[s];
    int M = Bx.M;
    dim3 gH(ntiles(Hh), mtiles(M));        // M x 512 outputs
    dim3 gA(ntiles(M),  mtiles(M));        // M x M adjacency

    // X = elu(kmer @ kW + kb)             [M,8420]x[8420,512]
    gemm_wmma<1,0,0,0><<<gH, 256, 0, stream>>>(Bx.kmer, Bx.kW, Bx.kb, nullptr, nullptr,
                                               X, M, Hh, KMERD, KMERD, Hh, Hh);
    // RES = X @ shW + shb
    gemm_wmma<0,0,0,0><<<gH, 256, 0, stream>>>(X, shW, shb, nullptr, nullptr,
                                               RES, M, Hh, Hh, Hh, Hh, Hh);
    // X = elu(RES)
    k_elu<<<(M * Hh + 255) / 256, 256, 0, stream>>>(RES, X, M * Hh);
    // T = tanh(X @ aW + ab)
    gemm_wmma<2,0,0,0><<<gH, 256, 0, stream>>>(X, Bx.aW, Bx.ab, nullptr, nullptr,
                                               T, M, Hh, Hh, Hh, Hh, Hh);
    // Wn = row L2 norms of T
    k_rownorm<<<(M + 7) / 8, 256, 0, stream>>>(T, Wn, M);
    // adj = (T @ T^T) / (w w^T)   — SYRK with cosine epilogue, straight to d_out
    gemm_wmma<0,0,1,1><<<gA, 256, 0, stream>>>(T, T, nullptr, nullptr, Wn,
                                               Bx.adj, M, M, Hh, Hh, Hh, M);
    // T = X @ g1W ; RES += adj @ T + g1b
    gemm_wmma<0,0,0,0><<<gH, 256, 0, stream>>>(X, g1W, nullptr, nullptr, nullptr,
                                               T, M, Hh, Hh, Hh, Hh, Hh);
    gemm_wmma<0,1,0,0><<<gH, 256, 0, stream>>>(Bx.adj, T, g1b, RES, nullptr,
                                               RES, M, Hh, M, M, Hh, Hh);
    // X = elu(RES) ; T = X @ g2W ; RES += adj @ T + g2b
    k_elu<<<(M * Hh + 255) / 256, 256, 0, stream>>>(RES, X, M * Hh);
    gemm_wmma<0,0,0,0><<<gH, 256, 0, stream>>>(X, g2W, nullptr, nullptr, nullptr,
                                               T, M, Hh, Hh, Hh, Hh, Hh);
    gemm_wmma<0,1,0,0><<<gH, 256, 0, stream>>>(Bx.adj, T, g2b, RES, nullptr,
                                               RES, M, Hh, M, M, Hh, Hh);
    // res32 = elu(RES)[idx]
    k_gather_elu<<<BB, Hh, 0, stream>>>(RES, Bx.idx, Bx.res32);
  }

  // ---------------- CNN branches ----------------
  k_cnn<<<dim3((64 * LAB + 255) / 256, BB), 256, 0, stream>>>(ab_am, c1cW, c1cb, C1, LAB);
  k_cnn<<<dim3((64 * LV  + 255) / 256, BB), 256, 0, stream>>>(v_am,  c2cW, c2cb, C2, LV);
  {
    dim3 g1(ntiles(Hh), mtiles(BB));
    gemm_wmma<0,0,0,0><<<g1, 256, 0, stream>>>(C1, c1lW, c1lb, nullptr, nullptr,
                                               ABFT, BB, Hh, 64 * LAB, 64 * LAB, Hh, Hh);
    gemm_wmma<0,0,0,0><<<g1, 256, 0, stream>>>(C2, c2lW, c2lb, nullptr, nullptr,
                                               VFT, BB, Hh, 64 * LV, 64 * LV, Hh, Hh);
  }

  // ---------------- attention fusion + output head ----------------
  k_att<<<BB, Hh, 0, stream>>>(V32, AB32, VFT, ABFT, a1W, a1b, a2W, a2b, PAIR);
  {
    dim3 g(ntiles(Hh), mtiles(BB));
    gemm_wmma<0,0,0,0><<<g, 256, 0, stream>>>(PAIR, oW, ob, nullptr, nullptr,
                                              OUTF, BB, Hh, 1024, 1024, Hh, Hh);
  }
  k_pred<<<BB, Hh, 0, stream>>>(OUTF, pW, pb, out);
}
